// NunchakuFluxAttention_1683627180466
// MI455X (gfx1250) — matrix operand order repaired and checked by reading, not compiled, not run on hardware
//
#include <hip/hip_runtime.h>
#include <stdint.h>

// ---------------- Types for WMMA fragments / TDM descriptors ----------------
typedef __bf16 v16bf __attribute__((ext_vector_type(16)));
typedef float  v8f   __attribute__((ext_vector_type(8)));
typedef unsigned int u32x4 __attribute__((ext_vector_type(4)));
typedef int i32x4 __attribute__((ext_vector_type(4)));
typedef int i32x8 __attribute__((ext_vector_type(8)));

union FragAB { u32x4 q[2]; v16bf v; };

#define HEADS 24
#define HD    128
#define DM    3072
#define RANK  32
#define S_IMG 2048
#define S_TXT 512
#define S_TOT 2560
#define NQKV  9216      // 3*DM
#define KAUG  3104      // DM + RANK

// TDM availability (device pass only; host pass takes fallback branch)
#if defined(__AMDGCN__) && defined(__has_builtin)
#  if __has_builtin(__builtin_amdgcn_tensor_load_to_lds)
#    define USE_TDM 1
#  endif
#endif
#ifndef USE_TDM
#  define USE_TDM 0
#endif

// fp32 -> bf16 bits, round-to-nearest-even
__device__ __forceinline__ unsigned short f2bf(float x) {
  unsigned int u = __float_as_uint(x);
  unsigned int r = (u + 0x7FFFu + ((u >> 16) & 1u)) >> 16;
  return (unsigned short)r;
}

__device__ __forceinline__ float rmax16(float v) {
#pragma unroll
  for (int m = 1; m < 16; m <<= 1) v = fmaxf(v, __shfl_xor(v, m, 32));
  return v;
}
__device__ __forceinline__ float rsum16(float v) {
#pragma unroll
  for (int m = 1; m < 16; m <<= 1) v += __shfl_xor(v, m, 32);
  return v;
}

#if USE_TDM
// Issue a TDM 2-D tile load: 16 rows (n) x 32 cols (k) of bf16 from W into LDS.
__device__ __forceinline__ void tdm_issue_b(const unsigned short* W, int nrow0, int k0,
                                            int K, int N, unsigned int lds_dst) {
  size_t ga = (size_t)W + ((size_t)nrow0 * K + k0) * 2;
  u32x4 g0;
  g0[0] = 1u;                                    // count=1, user descriptor
  g0[1] = lds_dst;                               // LDS byte address
  g0[2] = (unsigned int)(ga & 0xFFFFFFFFu);      // global_addr[31:0]
  g0[3] = (unsigned int)((ga >> 32) & 0x01FFFFFFu) | (2u << 30);  // [56:32] | type=2
  i32x8 g1;
  g1[0] = (int)(1u << 16);                                       // data_size = 2 bytes
  g1[1] = (int)(((unsigned)K & 0xFFFFu) << 16);                  // tensor_dim0 lo16
  g1[2] = (int)((((unsigned)K >> 16) & 0xFFFFu) |
                (((unsigned)N & 0xFFFFu) << 16));                // dim0 hi | dim1 lo
  g1[3] = (int)((((unsigned)N >> 16) & 0xFFFFu) | (32u << 16));  // dim1 hi | tile_dim0=32
  g1[4] = 16;                                                    // tile_dim1=16, tile_dim2=0
  g1[5] = (int)(unsigned)K;                                      // tensor_dim0_stride[31:0]
  g1[6] = 0;
  g1[7] = 0;
  i32x4 gz4 = {0, 0, 0, 0};
#  if __clang_major__ >= 23
  i32x8 gz8 = {0, 0, 0, 0, 0, 0, 0, 0};
  __builtin_amdgcn_tensor_load_to_lds(g0, g1, gz4, gz4, gz8, 0);
#  else
  __builtin_amdgcn_tensor_load_to_lds(g0, g1, gz4, gz4, 0);
#  endif
}
__device__ __forceinline__ void tdm_wait() {
#  if __has_builtin(__builtin_amdgcn_s_wait_tensorcnt)
  __builtin_amdgcn_s_wait_tensorcnt(0);
#  else
  asm volatile("s_wait_tensorcnt 0" ::: "memory");
#  endif
}
#endif

// ---------------- Weight pack: W' = [W | B_lr] cast to bf16 ----------------
__global__ void build_w_kernel(const float* __restrict__ w, const float* __restrict__ bmat,
                               unsigned short* __restrict__ wp, int N, int K, int R) {
  size_t total = (size_t)N * (K + R);
  for (size_t i = (size_t)blockIdx.x * blockDim.x + threadIdx.x; i < total;
       i += (size_t)gridDim.x * blockDim.x) {
    int kr = (int)(i % (size_t)(K + R));
    size_t n = i / (size_t)(K + R);
    float v = (kr < K) ? w[n * (size_t)K + kr] : bmat[n * (size_t)R + (kr - K)];
    wp[i] = f2bf(v);
  }
}

// ---------------- Activation pack: X' = [x | x @ a^T] cast to bf16 ----------------
__global__ __launch_bounds__(256) void build_x_kernel(const float* __restrict__ x,
                                                      const float* __restrict__ amat,
                                                      unsigned short* __restrict__ xp) {
  const int s = blockIdx.x;
  const float* xrow = x + (size_t)s * DM;
  unsigned short* orow = xp + (size_t)s * KAUG;
  for (int c = threadIdx.x; c < DM; c += 256) orow[c] = f2bf(xrow[c]);
  const int r = threadIdx.x & 31;       // rank index
  const int part = threadIdx.x >> 5;    // 8 K-partitions of 384
  float acc = 0.f;
  const float* arow = amat + (size_t)r * DM;
  for (int k = part * 384; k < part * 384 + 384; ++k) acc += xrow[k] * arow[k];
  __shared__ float ps[8][32];
  ps[part][r] = acc;
  __syncthreads();
  if (threadIdx.x < 32) {
    float t = 0.f;
#pragma unroll
    for (int p = 0; p < 8; ++p) t += ps[p][threadIdx.x];
    orow[DM + threadIdx.x] = f2bf(t);
  }
}

// ---------------- bf16 WMMA GEMM: C(MxN) = X(MxK) * W(NxK)^T + bias ----------------
// 256 thr = 8 waves, tile 128x128, BK=32, double-buffered LDS, software pipelined.
// B tile staged by the Tensor Data Mover (overlapped with WMMA); A via b128 loads.
__global__ __launch_bounds__(256) void gemm_bf16_wmma(const unsigned short* __restrict__ X,
                                                      const unsigned short* __restrict__ W,
                                                      const float* __restrict__ bias,
                                                      float* __restrict__ C,
                                                      int M, int N, int K) {
  __shared__ __align__(16) unsigned int Asu[2][128 * 16];  // [buf][128 m][32 bf16 k]
  __shared__ __align__(16) unsigned int Bsu[2][128 * 16];  // [buf][128 n][32 bf16 k]
  const int tid = threadIdx.x;
  const int lane = tid & 31, wave = tid >> 5;
  const int nl = lane & 15, kh = lane >> 4;
  const int wm = wave & 3, wn = wave >> 2;  // 4 x 2 wave grid
  const int m0 = blockIdx.y * 128, n0 = blockIdx.x * 128;
  const u32x4* X4 = (const u32x4*)X;
  const int Kq = K >> 3;  // u32x4 per row
  const int lin0 = tid, lin1 = tid + 256;   // 512 u32x4 per tile, 2 per thread
  const int r0 = lin0 >> 2, c0 = lin0 & 3;
  const int r1 = lin1 >> 2, c1 = lin1 & 3;
#if !USE_TDM
  const u32x4* W4 = (const u32x4*)W;
#else
  const int nrow0 = n0 + wave * 16;
#endif

  v8f acc[2][4];
#pragma unroll
  for (int a = 0; a < 2; ++a)
#pragma unroll
    for (int b = 0; b < 4; ++b)
#pragma unroll
      for (int j = 0; j < 8; ++j) acc[a][b][j] = 0.f;

  // ---- prologue: stage k0 = 0 into buffer 0 ----
  {
    *(u32x4*)&Asu[0][lin0 << 2] = X4[(size_t)(m0 + r0) * Kq + c0];
    *(u32x4*)&Asu[0][lin1 << 2] = X4[(size_t)(m0 + r1) * Kq + c1];
#if USE_TDM
    tdm_issue_b(W, nrow0, 0, K, N, (unsigned int)(size_t)&Bsu[0][wave * 256]);
    tdm_wait();
#else
    *(u32x4*)&Bsu[0][lin0 << 2] = W4[(size_t)(n0 + r0) * Kq + c0];
    *(u32x4*)&Bsu[0][lin1 << 2] = W4[(size_t)(n0 + r1) * Kq + c1];
#endif
  }
  __syncthreads();

  int buf = 0;
  for (int k0 = 0; k0 < K; k0 += 32) {
    const bool has_next = (k0 + 32) < K;
    u32x4 a0, a1;
#if !USE_TDM
    u32x4 b0r, b1r;
#endif
    if (has_next) {  // issue next-tile traffic before compute (overlap)
      const int kq = (k0 + 32) >> 3;
      a0 = X4[(size_t)(m0 + r0) * Kq + kq + c0];
      a1 = X4[(size_t)(m0 + r1) * Kq + kq + c1];
#if USE_TDM
      tdm_issue_b(W, nrow0, k0 + 32, K, N,
                  (unsigned int)(size_t)&Bsu[buf ^ 1][wave * 256]);
#else
      b0r = W4[(size_t)(n0 + r0) * Kq + kq + c0];
      b1r = W4[(size_t)(n0 + r1) * Kq + kq + c1];
#endif
    }

    // ---- compute from current buffer: 8 WMMAs ----
    const unsigned int* Ab = Asu[buf];
    const unsigned int* Bb = Bsu[buf];
    FragAB a[2], b[4];
#pragma unroll
    for (int mi = 0; mi < 2; ++mi) {  // A-layout: lane=row, K pairs split by lane-half
      int base = (wm * 32 + mi * 16 + nl) * 16;
      a[mi].q[0] = *(const u32x4*)(Ab + base + kh * 4);
      a[mi].q[1] = *(const u32x4*)(Ab + base + 8 + kh * 4);
    }
#pragma unroll
    for (int ni = 0; ni < 4; ++ni) {  // B-layout: lane=col n, K pairs split by lane-half
      int base = (wn * 64 + ni * 16 + nl) * 16;
      b[ni].q[0] = *(const u32x4*)(Bb + base + kh * 4);
      b[ni].q[1] = *(const u32x4*)(Bb + base + 8 + kh * 4);
    }
#pragma unroll
    for (int mi = 0; mi < 2; ++mi)
#pragma unroll
      for (int ni = 0; ni < 4; ++ni)
        acc[mi][ni] = __builtin_amdgcn_wmma_f32_16x16x32_bf16(
            false, a[mi].v, false, b[ni].v, (short)0, acc[mi][ni], false, false);

    if (has_next) {  // land next A tile; ensure next B tile arrived
      *(u32x4*)&Asu[buf ^ 1][lin0 << 2] = a0;
      *(u32x4*)&Asu[buf ^ 1][lin1 << 2] = a1;
#if USE_TDM
      tdm_wait();
#else
      *(u32x4*)&Bsu[buf ^ 1][lin0 << 2] = b0r;
      *(u32x4*)&Bsu[buf ^ 1][lin1 << 2] = b1r;
#endif
    }
    __syncthreads();
    buf ^= 1;
  }

#pragma unroll
  for (int mi = 0; mi < 2; ++mi)
#pragma unroll
    for (int ni = 0; ni < 4; ++ni)
#pragma unroll
      for (int j = 0; j < 8; ++j) {
        int r = m0 + wm * 32 + mi * 16 + j + 8 * kh;  // C layout: M=vgpr(+8), N=lane%16
        int c = n0 + wn * 64 + ni * 16 + nl;
        if (r < M) C[(size_t)r * N + c] = acc[mi][ni][j] + bias[c];
      }
}

// ---------------- RMSNorm + RoPE + scale, fp32 qkv -> bf16 q/k/v [h][s][d] ----------------
__global__ void norm_rope_kernel(const float* __restrict__ qkv, const float* __restrict__ nqw,
                                 const float* __restrict__ nkw, const float* __restrict__ naqw,
                                 const float* __restrict__ nakw, const float* __restrict__ cosb,
                                 const float* __restrict__ sinb, unsigned short* __restrict__ qb,
                                 unsigned short* __restrict__ kb, unsigned short* __restrict__ vb) {
  const int h = blockIdx.x, s = blockIdx.y, d = threadIdx.x;
  const float* row = qkv + (size_t)s * NQKV;
  float qv = row[h * HD + d];
  float kv = row[DM + h * HD + d];
  float vv = row[2 * DM + h * HD + d];
  const float* qw = (s < S_TXT) ? naqw : nqw;
  const float* kw = (s < S_TXT) ? nakw : nkw;
  __shared__ float red[HD];
  __shared__ float qs[HD];
  __shared__ float ks[HD];
  red[d] = qv * qv; __syncthreads();
  for (int off = 64; off > 0; off >>= 1) { if (d < off) red[d] += red[d + off]; __syncthreads(); }
  float qn = qv * rsqrtf(red[0] * (1.f / HD) + 1e-6f) * qw[d];
  __syncthreads();
  red[d] = kv * kv; __syncthreads();
  for (int off = 64; off > 0; off >>= 1) { if (d < off) red[d] += red[d + off]; __syncthreads(); }
  float kn = kv * rsqrtf(red[0] * (1.f / HD) + 1e-6f) * kw[d];
  qs[d] = qn; ks[d] = kn; __syncthreads();
  float c = cosb[(size_t)s * HD + d], sn = sinb[(size_t)s * HD + d];
  float qrot = (d & 1) ? qs[d - 1] : -qs[d + 1];
  float krot = (d & 1) ? ks[d - 1] : -ks[d + 1];
  float qo = (qn * c + qrot * sn) * 0.08838834764831845f;  // 1/sqrt(128) folded in
  float ko = kn * c + krot * sn;
  size_t idx = ((size_t)h * S_TOT + s) * HD + d;
  qb[idx] = f2bf(qo); kb[idx] = f2bf(ko); vb[idx] = f2bf(vv);
}

// ---------------- Flash attention, bf16 WMMA, online softmax ----------------
// grid: (S_TOT/128, HEADS); 8 waves, each owns 16 q-rows; kv streamed in steps of 32.
__global__ __launch_bounds__(256) void flash_attn_wmma(const unsigned short* __restrict__ qb,
                                                       const unsigned short* __restrict__ kb,
                                                       const unsigned short* __restrict__ vb,
                                                       unsigned short* __restrict__ attn_out) {
  __shared__ __align__(16) unsigned int Ktu[32 * 64];   // [kv row][128 bf16]
  __shared__ __align__(16) unsigned int Vtu[128 * 16];  // transposed: [d][32 bf16 kv]
  __shared__ __align__(16) unsigned int Pbu[8][16 * 16];// per-wave P tile [16 r][32 bf16]
  const int tid = threadIdx.x;
  const int lane = tid & 31, wave = tid >> 5;
  const int nl = lane & 15, kh = lane >> 4;
  const int h = blockIdx.y;
  const int q0 = blockIdx.x * 128;
  const unsigned int* qbu = (const unsigned int*)qb;
  const u32x4* kb4 = (const u32x4*)kb;
  const unsigned short* vbs = vb;

  FragAB qf[4];
  {
    size_t rowbase = ((size_t)h * S_TOT + (q0 + wave * 16 + nl)) * (HD / 2);
#pragma unroll
    for (int f = 0; f < 4; ++f) {
      qf[f].q[0] = *(const u32x4*)(qbu + rowbase + f * 16 + kh * 4);
      qf[f].q[1] = *(const u32x4*)(qbu + rowbase + f * 16 + 8 + kh * 4);
    }
  }

  v8f o[8];
#pragma unroll
  for (int dt = 0; dt < 8; ++dt)
#pragma unroll
    for (int j = 0; j < 8; ++j) o[dt][j] = 0.f;
  float mrow[8], lrow[8];
#pragma unroll
  for (int j = 0; j < 8; ++j) { mrow[j] = -3e38f; lrow[j] = 0.f; }

  for (int kv0 = 0; kv0 < S_TOT; kv0 += 32) {
    __syncthreads();
#pragma unroll
    for (int q = 0; q < 2; ++q) {  // K tile row-major: 512 u32x4, 2 per thread
      int lin4 = tid + q * 256;
      int r = lin4 >> 4, c4 = lin4 & 15;
      *(u32x4*)&Ktu[lin4 << 2] = kb4[((size_t)h * S_TOT + kv0 + r) * 16 + c4];
    }
#pragma unroll
    for (int q = 0; q < 8; ++q) {  // V tile transposed, packed k-pairs
      int lin = tid + q * 256;
      int d = lin >> 4, i = lin & 15;
      unsigned int lo = vbs[((size_t)h * S_TOT + kv0 + 2 * i) * HD + d];
      unsigned int hi = vbs[((size_t)h * S_TOT + kv0 + 2 * i + 1) * HD + d];
      Vtu[lin] = lo | (hi << 16);
    }
    __syncthreads();

    v8f s0, s1;
#pragma unroll
    for (int j = 0; j < 8; ++j) { s0[j] = 0.f; s1[j] = 0.f; }
#pragma unroll
    for (int c = 0; c < 4; ++c) {
      FragAB b0, b1;
      int base0 = nl * 64 + c * 16 + kh * 4;
      b0.q[0] = *(const u32x4*)(Ktu + base0);
      b0.q[1] = *(const u32x4*)(Ktu + base0 + 8);
      int base1 = (nl + 16) * 64 + c * 16 + kh * 4;
      b1.q[0] = *(const u32x4*)(Ktu + base1);
      b1.q[1] = *(const u32x4*)(Ktu + base1 + 8);
      s0 = __builtin_amdgcn_wmma_f32_16x16x32_bf16(false, qf[c].v, false, b0.v, (short)0, s0, false, false);
      s1 = __builtin_amdgcn_wmma_f32_16x16x32_bf16(false, qf[c].v, false, b1.v, (short)0, s1, false, false);
    }

    float p0[8], p1[8], scl[8];
#pragma unroll
    for (int j = 0; j < 8; ++j) {
      float t = rmax16(fmaxf(s0[j], s1[j]));
      float mn = fmaxf(mrow[j], t);
      scl[j] = __expf(mrow[j] - mn);
      mrow[j] = mn;
      p0[j] = __expf(s0[j] - mn);
      p1[j] = __expf(s1[j] - mn);
      lrow[j] = lrow[j] * scl[j] + rsum16(p0[j] + p1[j]);
    }
#pragma unroll
    for (int dt = 0; dt < 8; ++dt)
#pragma unroll
      for (int j = 0; j < 8; ++j) o[dt][j] *= scl[j];

    unsigned short* Pw = (unsigned short*)Pbu[wave];
#pragma unroll
    for (int j = 0; j < 8; ++j) {
      int row = j + 8 * kh;
      Pw[row * 32 + nl] = f2bf(p0[j]);
      Pw[row * 32 + 16 + nl] = f2bf(p1[j]);
    }
    asm volatile("s_wait_dscnt 0" ::: "memory");  // wave-local LDS store->load hazard

    FragAB pf;
    {
      const unsigned int* Pr = Pbu[wave];
      int base = nl * 16;
      pf.q[0] = *(const u32x4*)(Pr + base + kh * 4);
      pf.q[1] = *(const u32x4*)(Pr + base + 8 + kh * 4);
    }
#pragma unroll
    for (int dt = 0; dt < 8; ++dt) {
      FragAB vf;
      int base = (dt * 16 + nl) * 16 + kh * 4;
      vf.q[0] = *(const u32x4*)(Vtu + base);
      vf.q[1] = *(const u32x4*)(Vtu + base + 8);
      o[dt] = __builtin_amdgcn_wmma_f32_16x16x32_bf16(false, pf.v, false, vf.v, (short)0, o[dt], false, false);
    }
  }

#pragma unroll
  for (int dt = 0; dt < 8; ++dt)
#pragma unroll
    for (int j = 0; j < 8; ++j) {
      int r = q0 + wave * 16 + j + 8 * kh;
      int c = h * HD + dt * 16 + nl;
      attn_out[(size_t)r * DM + c] = f2bf(o[dt][j] / lrow[j]);
    }
}

// ---------------- Launch ----------------
extern "C" void kernel_launch(void* const* d_in, const int* in_sizes, int n_in,
                              void* d_out, int out_size, void* d_ws, size_t ws_size,
                              hipStream_t stream) {
  const float* hidden    = (const float*)d_in[0];
  const float* enc       = (const float*)d_in[1];
  const float* rope_cos  = (const float*)d_in[2];
  const float* rope_sin  = (const float*)d_in[3];
  const float* qkv_w     = (const float*)d_in[4];
  const float* qkv_a     = (const float*)d_in[5];
  const float* qkv_b     = (const float*)d_in[6];
  const float* qkv_bias  = (const float*)d_in[7];
  const float* aqkv_w    = (const float*)d_in[8];
  const float* aqkv_a    = (const float*)d_in[9];
  const float* aqkv_b    = (const float*)d_in[10];
  const float* aqkv_bias = (const float*)d_in[11];
  const float* norm_q_w  = (const float*)d_in[12];
  const float* norm_k_w  = (const float*)d_in[13];
  const float* norm_aq_w = (const float*)d_in[14];
  const float* norm_ak_w = (const float*)d_in[15];
  const float* out_w     = (const float*)d_in[16];
  const float* out_b     = (const float*)d_in[17];
  const float* aout_w    = (const float*)d_in[18];
  const float* aout_b    = (const float*)d_in[19];
  float* out = (float*)d_out;

  char* ws = (char*)d_ws;
  size_t off = 0;
  auto alloc = [&](size_t bytes) {
    size_t o = off;
    off = (off + bytes + 255) & ~(size_t)255;
    return o;
  };
  unsigned short* Xq  = (unsigned short*)(ws + alloc((size_t)S_IMG * KAUG * 2));
  unsigned short* Xe  = (unsigned short*)(ws + alloc((size_t)S_TXT * KAUG * 2));
  unsigned short* Wq  = (unsigned short*)(ws + alloc((size_t)NQKV * KAUG * 2));
  unsigned short* We  = (unsigned short*)(ws + alloc((size_t)NQKV * KAUG * 2));
  unsigned short* Wo  = (unsigned short*)(ws + alloc((size_t)DM * DM * 2));
  unsigned short* Wao = (unsigned short*)(ws + alloc((size_t)DM * DM * 2));
  float* qkvbuf       = (float*)(ws + alloc((size_t)S_TOT * NQKV * 4));
  unsigned short* qb  = (unsigned short*)(ws + alloc((size_t)HEADS * S_TOT * HD * 2));
  unsigned short* kb  = (unsigned short*)(ws + alloc((size_t)HEADS * S_TOT * HD * 2));
  unsigned short* vb  = (unsigned short*)(ws + alloc((size_t)HEADS * S_TOT * HD * 2));
  unsigned short* att = (unsigned short*)(ws + alloc((size_t)S_TOT * DM * 2));

  build_w_kernel<<<2048, 256, 0, stream>>>(qkv_w, qkv_b, Wq, NQKV, DM, RANK);
  build_w_kernel<<<2048, 256, 0, stream>>>(aqkv_w, aqkv_b, We, NQKV, DM, RANK);
  build_w_kernel<<<1024, 256, 0, stream>>>(out_w, out_w, Wo, DM, DM, 0);
  build_w_kernel<<<1024, 256, 0, stream>>>(aout_w, aout_w, Wao, DM, DM, 0);
  build_x_kernel<<<S_IMG, 256, 0, stream>>>(hidden, qkv_a, Xq);
  build_x_kernel<<<S_TXT, 256, 0, stream>>>(enc, aqkv_a, Xe);

  dim3 gq(NQKV / 128, S_IMG / 128);
  gemm_bf16_wmma<<<gq, 256, 0, stream>>>(Xq, Wq, qkv_bias, qkvbuf + (size_t)S_TXT * NQKV,
                                         S_IMG, NQKV, KAUG);
  dim3 ge(NQKV / 128, S_TXT / 128);
  gemm_bf16_wmma<<<ge, 256, 0, stream>>>(Xe, We, aqkv_bias, qkvbuf, S_TXT, NQKV, KAUG);

  dim3 gn(HEADS, S_TOT);
  norm_rope_kernel<<<gn, HD, 0, stream>>>(qkvbuf, norm_q_w, norm_k_w, norm_aq_w, norm_ak_w,
                                          rope_cos, rope_sin, qb, kb, vb);

  dim3 ga(S_TOT / 128, HEADS);
  flash_attn_wmma<<<ga, 256, 0, stream>>>(qb, kb, vb, att);

  dim3 go1(DM / 128, S_IMG / 128);
  gemm_bf16_wmma<<<go1, 256, 0, stream>>>(att + (size_t)S_TXT * DM, Wo, out_b, out,
                                          S_IMG, DM, DM);
  dim3 go2(DM / 128, S_TXT / 128);
  gemm_bf16_wmma<<<go2, 256, 0, stream>>>(att, Wao, aout_b, out + (size_t)S_IMG * DM,
                                          S_TXT, DM, DM);
  (void)in_sizes; (void)n_in; (void)out_size; (void)ws_size; (void)rope_cos; (void)rope_sin;
}
